// ResidualAttentionBlock_81140522156718
// MI455X (gfx1250) — matrix operand order repaired and checked by reading, not compiled, not run on hardware
//
#include <hip/hip_runtime.h>
#include <cstdint>
#include <cstddef>

typedef __bf16 bf16;
typedef __attribute__((ext_vector_type(16))) __bf16        v16bf;
typedef __attribute__((ext_vector_type(8)))  float         v8f;
typedef __attribute__((ext_vector_type(4)))  unsigned int  v4u;
typedef __attribute__((ext_vector_type(8)))  unsigned int  v8u;

union Frag {
  v16bf v;
  v4u   u[2];
};

__device__ __forceinline__ v8f wmma_bf16(v16bf a, v16bf b, v8f c) {
  // D = A(16x32) * B(32x16) + C, fp32 accumulate
  return __builtin_amdgcn_wmma_f32_16x16x32_bf16(
      /*neg_a=*/false, a, /*neg_b=*/false, b,
      /*c_mod=*/(short)0, c, /*reuse_a=*/false, /*reuse_b=*/false);
}

// Async global->LDS staging: INST_OFFSET applies to BOTH global and LDS
// addresses, so one (lds,global) base pair + offsets stages a contiguous run.
__device__ __forceinline__ void async_copy_64B(unsigned lds, unsigned long long ga) {
  asm volatile(
      "global_load_async_to_lds_b128 %0, %1, off\n\t"
      "global_load_async_to_lds_b128 %0, %1, off offset:16\n\t"
      "global_load_async_to_lds_b128 %0, %1, off offset:32\n\t"
      "global_load_async_to_lds_b128 %0, %1, off offset:48"
      :: "v"(lds), "v"(ga) : "memory");
}
__device__ __forceinline__ void async_copy_32B(unsigned lds, unsigned long long ga) {
  asm volatile(
      "global_load_async_to_lds_b128 %0, %1, off\n\t"
      "global_load_async_to_lds_b128 %0, %1, off offset:16"
      :: "v"(lds), "v"(ga) : "memory");
}
__device__ __forceinline__ void wait_async0() {
  asm volatile("s_wait_asynccnt 0x0" ::: "memory");
}

// ---------------------------------------------------------------------------
// fp32 -> bf16 elementwise convert (weights)
// ---------------------------------------------------------------------------
__global__ __launch_bounds__(256)
void cvt_f32_bf16(const float* __restrict__ in, bf16* __restrict__ out, int n) {
  int i = blockIdx.x * 256 + threadIdx.x;
  if (i < n) out[i] = (bf16)in[i];
}

// ---------------------------------------------------------------------------
// LayerNorm(x) -> bf16, one block per row, D = 1024
// ---------------------------------------------------------------------------
__global__ __launch_bounds__(256)
void ln_to_bf16(const float* __restrict__ x, const float* __restrict__ g,
                const float* __restrict__ beta, bf16* __restrict__ out) {
  const int D = 1024;
  const int row = blockIdx.x;
  const int t = threadIdx.x;
  __shared__ float rs[256], rq[256];
  const float* xr = x + (size_t)row * D;
  float v0 = xr[t], v1 = xr[t + 256], v2 = xr[t + 512], v3 = xr[t + 768];
  rs[t] = v0 + v1 + v2 + v3;
  rq[t] = v0 * v0 + v1 * v1 + v2 * v2 + v3 * v3;
  __syncthreads();
  for (int off = 128; off > 0; off >>= 1) {
    if (t < off) { rs[t] += rs[t + off]; rq[t] += rq[t + off]; }
    __syncthreads();
  }
  const float mean = rs[0] * (1.0f / D);
  const float var  = rq[0] * (1.0f / D) - mean * mean;
  const float rstd = rsqrtf(var + 1e-5f);
  bf16* orow = out + (size_t)row * D;
  orow[t]       = (bf16)((v0 - mean) * rstd * g[t]       + beta[t]);
  orow[t + 256] = (bf16)((v1 - mean) * rstd * g[t + 256] + beta[t + 256]);
  orow[t + 512] = (bf16)((v2 - mean) * rstd * g[t + 512] + beta[t + 512]);
  orow[t + 768] = (bf16)((v3 - mean) * rstd * g[t + 768] + beta[t + 768]);
}

// ---------------------------------------------------------------------------
// C[M,N] = epilogue( A[M,K](bf16) @ W[N,K]^T(bf16) + bias )
// Block tile 256x128, K-step 32, 8 waves (wave tile 64x64 = 16 WMMA/step).
// Tiles staged with global_load_async_to_lds_b128 (ASYNCcnt), double-buffered.
// EPI 0: Cf = acc + bias + resid                       (fp32 out)
// EPI 1: Cb = bf16( (acc+bias) * (col<qn ? qs : 1) )   (QKV, q pre-scaled)
// EPI 2: Cb = bf16( quickgelu(acc+bias) )
// M multiple of 256, N multiple of 128, K multiple of 32.
// ---------------------------------------------------------------------------
template <int EPI>
__global__ __launch_bounds__(256)
void gemm_bf16_wmma(const bf16* __restrict__ A, const bf16* __restrict__ W,
                    const float* __restrict__ bias, const float* __restrict__ resid,
                    float* __restrict__ Cf, bf16* __restrict__ Cb,
                    int M, int N, int K, float qs, int qn) {
  __shared__ bf16 sA[2][256][40];   // stride 40 elems = 80 B (16B-aligned rows)
  __shared__ bf16 sB[2][128][40];

  const int t    = threadIdx.x;
  const int lane = t & 31;
  const int lm   = lane & 15;
  const int hi   = (lane < 16) ? 0 : 1;
  const int wid  = t >> 5;
  const int wm   = (wid & 3) * 64;   // wave M offset in block tile
  const int wn   = (wid >> 2) * 64;  // wave N offset in block tile

  // A loader: thread t owns row t (0..255), 64 B per K-step
  // B loader: thread t owns row t>>1 (0..127), 32 B at col (t&1)*16
  const size_t aRow = (size_t)(blockIdx.y * 256 + t) * K;
  const int    brow = t >> 1;
  const int    bcol = (t & 1) << 4;
  const size_t bRow = (size_t)(blockIdx.x * 128 + brow) * K + bcol;

  v8f acc[4][4] = {};

  auto stage = [&](int buf, int kt) {
    const size_t ko = (size_t)kt * 32;
    async_copy_64B((unsigned)(uintptr_t)&sA[buf][t][0],
                   (unsigned long long)(uintptr_t)(A + aRow + ko));
    async_copy_32B((unsigned)(uintptr_t)&sB[buf][brow][bcol],
                   (unsigned long long)(uintptr_t)(W + bRow + ko));
  };

  stage(0, 0);
  wait_async0();
  __syncthreads();

  const int nk = K >> 5;
  const int ka = hi ? 8 : 0;    // A-fragment K half select (16-bit A layout)
  const int kb = hi ? 8 : 0;    // B-fragment K half select

  for (int kt = 0; kt < nk; ++kt) {
    const int cur = kt & 1;
    const bool pf = (kt + 1) < nk;
    if (pf) stage(cur ^ 1, kt + 1);

    Frag af[4], bfr[4];
    for (int mi = 0; mi < 4; ++mi) {
      const bf16* p = &sA[cur][wm + mi * 16 + lm][ka];
      af[mi].u[0] = *(const v4u*)p;          // K = ka .. ka+7
      af[mi].u[1] = *(const v4u*)(p + 16);   // K = 16+ka .. 16+ka+7
    }
    for (int ni = 0; ni < 4; ++ni) {
      const bf16* p = &sB[cur][wn + ni * 16 + lm][kb];
      bfr[ni].u[0] = *(const v4u*)p;         // K = kb .. kb+7
      bfr[ni].u[1] = *(const v4u*)(p + 16);  // K = 16+kb .. 16+kb+7
    }
    for (int mi = 0; mi < 4; ++mi)
      for (int ni = 0; ni < 4; ++ni)
        acc[mi][ni] = wmma_bf16(af[mi].v, bfr[ni].v, acc[mi][ni]);

    if (pf) wait_async0();
    __syncthreads();
  }

  // epilogue: C layout — VGPR i holds row (i + hi*8), col = lm within 16x16 tile
  const int rowBase = blockIdx.y * 256 + wm + hi * 8;
  const int colBase = blockIdx.x * 128 + wn + lm;
  for (int mi = 0; mi < 4; ++mi) {
    for (int ni = 0; ni < 4; ++ni) {
      for (int i = 0; i < 8; ++i) {
        const int row = rowBase + mi * 16 + i;
        const int col = colBase + ni * 16;
        const size_t idx = (size_t)row * N + col;
        float v = acc[mi][ni][i] + bias[col];
        if (EPI == 0) {
          Cf[idx] = v + resid[idx];
        } else if (EPI == 1) {
          if (col < qn) v *= qs;
          Cb[idx] = (bf16)v;
        } else {
          v = v * (1.0f / (1.0f + __expf(-1.702f * v)));
          Cb[idx] = (bf16)v;
        }
      }
    }
  }
}

// ---------------------------------------------------------------------------
// Flash attention: grid(16 qblocks, 64 bh), 128 threads = 4 waves.
// K tiles DMA'd into LDS by the Tensor Data Mover (2D D#, LDS row padding
// 128B data + 16B pad -> 72-elem stride), V transposed manually; online
// softmax; all matmuls via v_wmma_f32_16x16x32_bf16.
// qkv: [B*S, 3*D] bf16, q already scaled by hd^-0.5. o: [B*S, D] bf16.
// ---------------------------------------------------------------------------
__global__ __launch_bounds__(128)
void attn_fmha(const bf16* __restrict__ qkv, bf16* __restrict__ o) {
  const int S = 1024, D3 = 3072, HD = 64;
  const int qb = blockIdx.x;
  const int b  = blockIdx.y >> 4;
  const int h  = blockIdx.y & 15;
  const int t    = threadIdx.x;
  const int lane = t & 31;
  const int w    = t >> 5;
  const int lm   = lane & 15;
  const int hi   = (lane < 16) ? 0 : 1;

  __shared__ bf16  sK [64][72];   // [key][hd]   (filled by TDM)
  __shared__ bf16  sVt[64][72];   // [hd][key]   (transposed V)
  __shared__ bf16  sP [64][72];   // [q][key]    softmax probabilities
  __shared__ float sS[4][16][64]; // per-wave raw scores
  __shared__ float sM[64], sL[64], sC[64];

  const size_t rowB = (size_t)b * S;
  const int hb = h * HD;

  if (t < 64) { sM[t] = -1e30f; sL[t] = 0.0f; }

  // Q fragments for this wave's 16 rows (K-dim = hd = 64 -> two 16x32 frags)
  const int qg = qb * 64 + w * 16 + lm;
  const bf16* qp = qkv + (rowB + qg) * D3 + hb;
  const int ka = hi ? 8 : 0;
  const int kb = hi ? 8 : 0;
  Frag aq[2];
  aq[0].u[0] = *(const v4u*)(qp + ka);
  aq[0].u[1] = *(const v4u*)(qp + 16 + ka);
  aq[1].u[0] = *(const v4u*)(qp + 32 + ka);
  aq[1].u[1] = *(const v4u*)(qp + 48 + ka);

  v8f oacc[4] = {};

  const int kr = t >> 1;          // loader key row 0..63
  const int c0 = (t & 1) << 5;    // loader hd col 0 or 32

  for (int kt = 0; kt < 16; ++kt) {
    const int k0 = kt * 64;

    // --- K tile via Tensor Data Mover: 64x64 bf16 tile, row stride 3072,
    //     LDS padding: 32 DWORDs data + 4 DWORDs pad per row.
    if (w == 0) {
      const unsigned long long ga =
          (unsigned long long)(uintptr_t)(qkv + (rowB + k0) * D3 + 1024 + hb);
      v4u g0;
      g0[0] = 1u;                                   // count = 1 descriptor
      g0[1] = (unsigned)(uintptr_t)&sK[0][0];       // lds_addr (bytes)
      g0[2] = (unsigned)ga;                         // global_addr[31:0]
      g0[3] = (unsigned)(ga >> 32) | (2u << 30);    // global_addr[56:32] | type=2
      v8u g1;
      g1[0] = (1u << 16)      // data_size = 2 bytes
            | (1u << 20)      // pad_enable
            | (4u << 22)      // pad_interval: 32 DWORDs (128 B)
            | (3u << 25);     // pad_amount:   4 DWORDs (16 B)
      g1[1] = (64u << 16);    // tensor_dim0 = 64 (low 16 bits @ bit 48)
      g1[2] = (64u << 16);    // tensor_dim0 hi=0 | tensor_dim1 = 64 (low 16)
      g1[3] = (64u << 16);    // tensor_dim1 hi=0 | tile_dim0 = 64
      g1[4] = 64u;            // tile_dim1 = 64, tile_dim2 = 0
      g1[5] = 3072u;          // tensor_dim0_stride[31:0]
      g1[6] = 0u;             // stride hi, dim1_stride lo
      g1[7] = 0u;             // dim1_stride hi
      asm volatile("tensor_load_to_lds %0, %1" :: "s"(g0), "s"(g1) : "memory");
    }

    // --- V tile (transposed into sVt), cooperative ---
    {
      const bf16* vp = qkv + (rowB + k0 + kr) * D3 + 2048 + hb + c0;
      union { v4u u; bf16 e[8]; } tv;
      for (int j = 0; j < 4; ++j) {
        tv.u = *(const v4u*)(vp + j * 8);
        for (int e = 0; e < 8; ++e) sVt[c0 + j * 8 + e][kr] = tv.e[e];
      }
    }
    if (w == 0) __builtin_amdgcn_s_wait_tensorcnt(0);
    __syncthreads();

    // --- S = Q @ K^T (16 q x 64 keys per wave) ---
    for (int ni = 0; ni < 4; ++ni) {
      v8f s = {};
      for (int kk = 0; kk < 2; ++kk) {
        Frag bk;
        const bf16* p = &sK[ni * 16 + lm][kk * 32 + kb];
        bk.u[0] = *(const v4u*)p;
        bk.u[1] = *(const v4u*)(p + 16);
        s = wmma_bf16(aq[kk].v, bk.v, s);
      }
      for (int i = 0; i < 8; ++i)
        sS[w][i + hi * 8][ni * 16 + lm] = s[i];
    }
    __syncthreads();

    // --- online softmax, one query row per lane (lanes 0..15 of each wave) ---
    if (lane < 16) {
      const int rl = w * 16 + lane;
      float m_old = sM[rl];
      float rmax = m_old;
      for (int j = 0; j < 64; ++j) rmax = fmaxf(rmax, sS[w][lane][j]);
      const float corr = __expf(m_old - rmax);
      float l = sL[rl] * corr;
      for (int j = 0; j < 64; ++j) {
        const float p = __expf(sS[w][lane][j] - rmax);
        l += p;
        sP[rl][j] = (bf16)p;
      }
      sM[rl] = rmax; sL[rl] = l; sC[rl] = corr;
    }
    __syncthreads();

    // --- O = diag(corr) * O + P @ V ---
    Frag ap[2];
    const bf16* pp = &sP[w * 16 + lm][0];
    ap[0].u[0] = *(const v4u*)(pp + ka);
    ap[0].u[1] = *(const v4u*)(pp + 16 + ka);
    ap[1].u[0] = *(const v4u*)(pp + 32 + ka);
    ap[1].u[1] = *(const v4u*)(pp + 48 + ka);
    for (int ni = 0; ni < 4; ++ni) {
      for (int i = 0; i < 8; ++i) oacc[ni][i] *= sC[w * 16 + hi * 8 + i];
      for (int kk = 0; kk < 2; ++kk) {
        Frag bv;
        const bf16* p = &sVt[ni * 16 + lm][kk * 32 + kb];
        bv.u[0] = *(const v4u*)p;
        bv.u[1] = *(const v4u*)(p + 16);
        oacc[ni] = wmma_bf16(ap[kk].v, bv.v, oacc[ni]);
      }
    }
    __syncthreads();
  }

  // --- normalize and store O (token-major [B*S, H*HD]) ---
  for (int ni = 0; ni < 4; ++ni) {
    for (int i = 0; i < 8; ++i) {
      const int rl = w * 16 + hi * 8 + i;
      const float inv = 1.0f / sL[rl];
      const int qrow = qb * 64 + rl;
      o[(rowB + qrow) * 1024 + hb + ni * 16 + lm] = (bf16)(oacc[ni][i] * inv);
    }
  }
}

// ---------------------------------------------------------------------------
// Launch
// ---------------------------------------------------------------------------
extern "C" void kernel_launch(void* const* d_in, const int* in_sizes, int n_in,
                              void* d_out, int out_size, void* d_ws, size_t ws_size,
                              hipStream_t stream) {
  (void)in_sizes; (void)n_in; (void)out_size; (void)ws_size;
  const int D = 1024, BS = 4096;

  const float* x   = (const float*)d_in[0];
  const float* ipw = (const float*)d_in[1];
  const float* ipb = (const float*)d_in[2];
  const float* ow  = (const float*)d_in[3];
  const float* ob  = (const float*)d_in[4];
  const float* g1  = (const float*)d_in[5];
  const float* b1  = (const float*)d_in[6];
  const float* fw  = (const float*)d_in[7];
  const float* fb  = (const float*)d_in[8];
  const float* pw  = (const float*)d_in[9];
  const float* pb  = (const float*)d_in[10];
  const float* g2  = (const float*)d_in[11];
  const float* b2  = (const float*)d_in[12];
  float* out = (float*)d_out;

  uintptr_t pws = (uintptr_t)d_ws;
  auto carve = [&](size_t bytes) -> void* {
    uintptr_t r = (pws + 255) & ~(uintptr_t)255;
    pws = r + bytes;
    return (void*)r;
  };
  bf16*  h1   = (bf16*)carve((size_t)BS * D * 2);
  bf16*  ipwB = (bf16*)carve((size_t)3 * D * D * 2);
  bf16*  owB  = (bf16*)carve((size_t)D * D * 2);
  bf16*  fwB  = (bf16*)carve((size_t)4 * D * D * 2);
  bf16*  pwB  = (bf16*)carve((size_t)4 * D * D * 2);
  bf16*  qkvB = (bf16*)carve((size_t)BS * 3 * D * 2);
  bf16*  oB   = (bf16*)carve((size_t)BS * D * 2);
  float* x2   = (float*)carve((size_t)BS * D * 4);
  bf16*  h2   = (bf16*)carve((size_t)BS * D * 2);
  bf16*  uB   = (bf16*)carve((size_t)BS * 4 * D * 2);

  // weights -> bf16
  cvt_f32_bf16<<<(3 * D * D + 255) / 256, 256, 0, stream>>>(ipw, ipwB, 3 * D * D);
  cvt_f32_bf16<<<(D * D + 255) / 256, 256, 0, stream>>>(ow, owB, D * D);
  cvt_f32_bf16<<<(4 * D * D + 255) / 256, 256, 0, stream>>>(fw, fwB, 4 * D * D);
  cvt_f32_bf16<<<(4 * D * D + 255) / 256, 256, 0, stream>>>(pw, pwB, 4 * D * D);

  // attention sublayer
  ln_to_bf16<<<BS, 256, 0, stream>>>(x, g1, b1, h1);
  gemm_bf16_wmma<1><<<dim3(3 * D / 128, BS / 256), 256, 0, stream>>>(
      h1, ipwB, ipb, nullptr, nullptr, qkvB, BS, 3 * D, D, 0.125f, D);
  attn_fmha<<<dim3(16, 64), 128, 0, stream>>>(qkvB, oB);
  gemm_bf16_wmma<0><<<dim3(D / 128, BS / 256), 256, 0, stream>>>(
      oB, owB, ob, x, x2, nullptr, BS, D, D, 0.0f, 0);

  // MLP sublayer
  ln_to_bf16<<<BS, 256, 0, stream>>>(x2, g2, b2, h2);
  gemm_bf16_wmma<2><<<dim3(4 * D / 128, BS / 256), 256, 0, stream>>>(
      h2, fwB, fb, nullptr, nullptr, uB, BS, 4 * D, D, 0.0f, 0);
  gemm_bf16_wmma<0><<<dim3(D / 128, BS / 256), 256, 0, stream>>>(
      uB, pwB, pb, x2, out, nullptr, BS, D, 4 * D, 0.0f, 0);
}
